// GRULayer_73478300500359
// MI455X (gfx1250) — matrix operand order repaired and checked by reading, not compiled, not run on hardware
//
#include <hip/hip_runtime.h>

// Problem dims
#define BB 64
#define TT 512
#define FF 256
#define HH 512
#define NWG 16            // persistent workgroups in the scan kernel
#define NCOL (HH / NWG)   // 32 output columns per workgroup
#define LDS_STRIDE (HH + 8)  // pad 16B per row -> conflict-free ds_load_b128

typedef _Float16 v8h  __attribute__((ext_vector_type(8)));
typedef _Float16 v16h __attribute__((ext_vector_type(16)));
typedef float    v8f  __attribute__((ext_vector_type(8)));
typedef int      v4i  __attribute__((ext_vector_type(4)));

// ---------------------------------------------------------------------------
// Address-space helpers (for the async global->LDS builtin).
// AS1 = global (64-bit VA); AS3 = LDS (32-bit offset space on gfx1250).
// ---------------------------------------------------------------------------
typedef __attribute__((address_space(1))) v4i gbl_v4i;
typedef __attribute__((address_space(3))) v4i lds_v4i;

__device__ __forceinline__ gbl_v4i* as_gbl_v4i(const void* p) {
  return (gbl_v4i*)(unsigned long long)p;
}
__device__ __forceinline__ lds_v4i* as_lds_v4i(void* p) {
  // generic LDS pointer low 32 bits == LDS offset
  return (lds_v4i*)(unsigned int)(unsigned long long)p;
}

// ---------------------------------------------------------------------------
// WMMA fragment loaders (wave32, v_wmma_f32_16x16x32_f16)
// A 16x32 (MxK) f16: lane L row M=L%16; elems 0..7 -> K=k0+half*8+e,
//                    elems 8..15 -> K=k0+16+half*8+(e-8).
// B 32x16 (KxN) f16: lane L col N=L%16; same K order as A.
// C/D 16x16 f32: elem v of lane L -> (M = v + 8*half, N = L%16).
// ---------------------------------------------------------------------------
__device__ __forceinline__ v16h load_a_frag(const _Float16* __restrict__ row,
                                            int k0, int halfsel) {
  const v8h lo = *(const v8h*)(row + k0 + halfsel * 8);
  const v8h hi = *(const v8h*)(row + k0 + 16 + halfsel * 8);
  return __builtin_shufflevector(lo, hi, 0, 1, 2, 3, 4, 5, 6, 7,
                                 8, 9, 10, 11, 12, 13, 14, 15);
}

__device__ __forceinline__ v16h load_b_frag(const _Float16* __restrict__ col,
                                            int k0, int halfsel) {
  const v8h lo = *(const v8h*)(col + k0 + halfsel * 8);
  const v8h hi = *(const v8h*)(col + k0 + 16 + halfsel * 8);
  return __builtin_shufflevector(lo, hi, 0, 1, 2, 3, 4, 5, 6, 7,
                                 8, 9, 10, 11, 12, 13, 14, 15);
}

__device__ __forceinline__ v8f wmma_f16(v16h a, v16h b, v8f c) {
  return __builtin_amdgcn_wmma_f32_16x16x32_f16(
      false, a, false, b, (short)0, c, false, false);
}

__device__ __forceinline__ float sigmoidf_fast(float x) {
  return 1.0f / (1.0f + __expf(-x));
}

// Stage one weight-slice (NCOL columns x HH halfs) into padded LDS rows.
__device__ __forceinline__ void stage_w(const _Float16* __restrict__ src,
                                        _Float16* dstbase, int col0, int tid) {
  for (int c = tid; c < NCOL * (HH / 8); c += 256) {
    const int col = c >> 6;          // HH/8 == 64 chunks per column
    const int kc = c & 63;
    const _Float16* s = src + (size_t)(col0 + col) * HH + kc * 8;
    _Float16* d = dstbase + col * LDS_STRIDE + kc * 8;
#if __has_builtin(__builtin_amdgcn_global_load_async_to_lds_b128)
    __builtin_amdgcn_global_load_async_to_lds_b128(as_gbl_v4i(s),
                                                   as_lds_v4i(d), 0, 0);
#else
    *(v8h*)d = *(const v8h*)s;
#endif
  }
}

__device__ __forceinline__ void wait_async_done() {
#if __has_builtin(__builtin_amdgcn_global_load_async_to_lds_b128)
#if __has_builtin(__builtin_amdgcn_s_wait_asynccnt)
  __builtin_amdgcn_s_wait_asynccnt(0);
#else
  asm volatile("s_wait_asynccnt 0" ::: "memory");
#endif
#endif
}

// ---------------------------------------------------------------------------
// Prep kernels
// ---------------------------------------------------------------------------
__global__ void convert_f32_to_f16(const float* __restrict__ src,
                                   _Float16* __restrict__ dst, int n) {
  int i = blockIdx.x * blockDim.x + threadIdx.x;
  if (i < n) dst[i] = (_Float16)src[i];
}

// dst is [N][K] row-major (W^T); src is [K][N] row-major
__global__ void transpose_w_to_f16(const float* __restrict__ src,
                                   _Float16* __restrict__ dst, int K, int N) {
  int idx = blockIdx.x * blockDim.x + threadIdx.x;
  if (idx < K * N) {
    int n = idx / K;
    int k = idx - n * K;
    dst[idx] = (_Float16)src[k * N + n];
  }
}

__global__ void init_state(_Float16* __restrict__ h, unsigned* __restrict__ ctr,
                           int n) {
  int i = blockIdx.x * blockDim.x + threadIdx.x;
  if (i < n) h[i] = (_Float16)0.0f;
  if (i == 0) *ctr = 0u;
}

// ---------------------------------------------------------------------------
// Persistent GRU scan. NWG blocks x 256 threads (8 wave32).
// Each block owns NCOL=32 output columns; each wave owns one 16x16 (m,n) tile.
// W_h slices live in LDS; W_i fragments get register-hoisted by the compiler;
// h state ping-pongs in global f16; one monotonic barrier per step, with the
// next step's x-path GEMM computed between barrier-arrive and barrier-wait.
// ---------------------------------------------------------------------------
__global__ __launch_bounds__(256) void gru_scan(
    const _Float16* __restrict__ xh,    // [B][T][F] f16
    const _Float16* __restrict__ wirT, const _Float16* __restrict__ wizT,
    const _Float16* __restrict__ winT,  // [H][F]
    const _Float16* __restrict__ whrT, const _Float16* __restrict__ whzT,
    const _Float16* __restrict__ whnT,  // [H][H]
    const float* __restrict__ b_ir, const float* __restrict__ b_iz,
    const float* __restrict__ b_in, const float* __restrict__ b_hn,
    _Float16* __restrict__ hbuf,        // [2][B][H] ping-pong
    unsigned* __restrict__ ctr,         // monotonic barrier counter
    float* __restrict__ out)            // [B][T][H] f32
{
  __shared__ _Float16 ldsW[3][NCOL][LDS_STRIDE];   // ~97.5 KB

  const int tid = threadIdx.x;
  const int wave = tid >> 5;
  const int lane = tid & 31;
  const int halfsel = lane >> 4;
  const int lm = lane & 15;

  const int m0 = (wave & 3) * 16;            // tile row base (batch)
  const int ncl = (wave >> 2) * 16 + lm;     // local column 0..31
  const int col0 = blockIdx.x * NCOL;
  const int mr = m0 + lm;                    // A-fragment row for this lane
  const int nc = col0 + ncl;                 // global output column

  // ---- stage W_h slices into LDS (async when available) ----
  stage_w(whrT, &ldsW[0][0][0], col0, tid);
  stage_w(whzT, &ldsW[1][0][0], col0, tid);
  stage_w(whnT, &ldsW[2][0][0], col0, tid);
  wait_async_done();
  __syncthreads();

  const _Float16* __restrict__ xrow0 = xh + (size_t)mr * TT * FF;
  const _Float16* __restrict__ wir_c = wirT + (size_t)nc * FF;
  const _Float16* __restrict__ wiz_c = wizT + (size_t)nc * FF;
  const _Float16* __restrict__ win_c = winT + (size_t)nc * FF;
  const _Float16* whr_l = &ldsW[0][ncl][0];
  const _Float16* whz_l = &ldsW[1][ncl][0];
  const _Float16* whn_l = &ldsW[2][ncl][0];

  const float bir = b_ir[nc];
  const float biz = b_iz[nc];
  const float bin = b_in[nc];
  const float bhn = b_hn[nc];

  // h(t-1) at this lane's C-fragment positions lives in registers.
  float hreg[8];
#pragma unroll
  for (int v = 0; v < 8; ++v) hreg[v] = 0.0f;

  // x-path accumulators for the *current* step (prologue computes t=0)
  v8f xr{}, xz{}, xn{};
  {
    const _Float16* xrow = xrow0;   // t = 0
#pragma unroll
    for (int k0 = 0; k0 < FF; k0 += 32) {
      v16h a = load_a_frag(xrow, k0, halfsel);
      xr = wmma_f16(a, load_b_frag(wir_c, k0, halfsel), xr);
      xz = wmma_f16(a, load_b_frag(wiz_c, k0, halfsel), xz);
      xn = wmma_f16(a, load_b_frag(win_c, k0, halfsel), xn);
    }
  }

  for (int t = 0; t < TT; ++t) {
    const _Float16* __restrict__ hprev = hbuf + (size_t)(t & 1) * BB * HH;
    _Float16* __restrict__ hnext = hbuf + (size_t)((t + 1) & 1) * BB * HH;

    // ---- hidden-path GEMM: h_{t-1} @ W_h{r,z,n}, K = H = 512 (B from LDS) --
    v8f acc_r = xr, acc_z = xz, acc_nh{};
    const _Float16* __restrict__ hrow = hprev + (size_t)mr * HH;
#pragma unroll 4
    for (int k0 = 0; k0 < HH; k0 += 32) {
      v16h a = load_a_frag(hrow, k0, halfsel);
      acc_r  = wmma_f16(a, load_b_frag(whr_l, k0, halfsel), acc_r);
      acc_z  = wmma_f16(a, load_b_frag(whz_l, k0, halfsel), acc_z);
      acc_nh = wmma_f16(a, load_b_frag(whn_l, k0, halfsel), acc_nh);
    }

    // ---- elementwise GRU update + stores ----
#pragma unroll
    for (int v = 0; v < 8; ++v) {
      const int m = m0 + v + 8 * halfsel;
      float r = sigmoidf_fast(acc_r[v] + bir);
      float z = sigmoidf_fast(acc_z[v] + biz);
      float nn = tanhf(xn[v] + bin + r * (acc_nh[v] + bhn));
      float hnew = (1.0f - z) * nn + z * hreg[v];
      _Float16 h16 = (_Float16)hnew;
      hreg[v] = (float)h16;   // keep register state consistent with f16 buffer
      hnext[(size_t)m * HH + nc] = h16;
      out[((size_t)m * TT + t) * HH + nc] = hnew;
    }

    // ---- barrier: arrive early, overlap next step's x-path, wait late ----
    __threadfence();   // release h_{t} stores to device scope
    __syncthreads();
    if (tid == 0) {
      __hip_atomic_fetch_add(ctr, 1u, __ATOMIC_RELEASE,
                             __HIP_MEMORY_SCOPE_AGENT);
    }

    if (t + 1 < TT) {  // uniform branch: x-path for step t+1 (h-independent)
      const _Float16* xrow = xrow0 + (size_t)(t + 1) * FF;
      v8f nxr{}, nxz{}, nxn{};
#pragma unroll
      for (int k0 = 0; k0 < FF; k0 += 32) {
        v16h a = load_a_frag(xrow, k0, halfsel);
        nxr = wmma_f16(a, load_b_frag(wir_c, k0, halfsel), nxr);
        nxz = wmma_f16(a, load_b_frag(wiz_c, k0, halfsel), nxz);
        nxn = wmma_f16(a, load_b_frag(win_c, k0, halfsel), nxn);
      }
      xr = nxr; xz = nxz; xn = nxn;
    }

    if (tid == 0) {
      const unsigned target = (unsigned)(t + 1) * NWG;
      while (__hip_atomic_load(ctr, __ATOMIC_ACQUIRE,
                               __HIP_MEMORY_SCOPE_AGENT) < target) {
        __builtin_amdgcn_s_sleep(1);
      }
    }
    __syncthreads();
    __threadfence();   // acquire other WGs' h_{t} stores
  }
}

// ---------------------------------------------------------------------------
// Host launcher
// ---------------------------------------------------------------------------
extern "C" void kernel_launch(void* const* d_in, const int* in_sizes, int n_in,
                              void* d_out, int out_size, void* d_ws,
                              size_t ws_size, hipStream_t stream) {
  (void)in_sizes; (void)n_in; (void)out_size; (void)ws_size;

  const float* x    = (const float*)d_in[0];
  const float* W_ir = (const float*)d_in[1];
  const float* W_iz = (const float*)d_in[2];
  const float* W_in = (const float*)d_in[3];
  const float* b_ir = (const float*)d_in[4];
  const float* b_iz = (const float*)d_in[5];
  const float* b_in = (const float*)d_in[6];
  const float* W_hr = (const float*)d_in[7];
  const float* W_hz = (const float*)d_in[8];
  const float* W_hn = (const float*)d_in[9];
  const float* b_hn = (const float*)d_in[10];
  float* out = (float*)d_out;

  char* ws = (char*)d_ws;
  size_t off = 0;
  _Float16* xh   = (_Float16*)(ws + off); off += (size_t)BB * TT * FF * 2;
  _Float16* wirT = (_Float16*)(ws + off); off += (size_t)HH * FF * 2;
  _Float16* wizT = (_Float16*)(ws + off); off += (size_t)HH * FF * 2;
  _Float16* winT = (_Float16*)(ws + off); off += (size_t)HH * FF * 2;
  _Float16* whrT = (_Float16*)(ws + off); off += (size_t)HH * HH * 2;
  _Float16* whzT = (_Float16*)(ws + off); off += (size_t)HH * HH * 2;
  _Float16* whnT = (_Float16*)(ws + off); off += (size_t)HH * HH * 2;
  _Float16* hbuf = (_Float16*)(ws + off); off += (size_t)2 * BB * HH * 2;
  unsigned* ctr  = (unsigned*)(ws + off); off += 256;

  const int nx = BB * TT * FF;
  convert_f32_to_f16<<<(nx + 255) / 256, 256, 0, stream>>>(x, xh, nx);

  const int nwi = FF * HH;
  const int nwh = HH * HH;
  transpose_w_to_f16<<<(nwi + 255) / 256, 256, 0, stream>>>(W_ir, wirT, FF, HH);
  transpose_w_to_f16<<<(nwi + 255) / 256, 256, 0, stream>>>(W_iz, wizT, FF, HH);
  transpose_w_to_f16<<<(nwi + 255) / 256, 256, 0, stream>>>(W_in, winT, FF, HH);
  transpose_w_to_f16<<<(nwh + 255) / 256, 256, 0, stream>>>(W_hr, whrT, HH, HH);
  transpose_w_to_f16<<<(nwh + 255) / 256, 256, 0, stream>>>(W_hz, whzT, HH, HH);
  transpose_w_to_f16<<<(nwh + 255) / 256, 256, 0, stream>>>(W_hn, whnT, HH, HH);

  const int nh = 2 * BB * HH;
  init_state<<<(nh + 255) / 256, 256, 0, stream>>>(hbuf, ctr, nh);

  gru_scan<<<NWG, 256, 0, stream>>>(xh, wirT, wizT, winT, whrT, whzT, whnT,
                                    b_ir, b_iz, b_in, b_hn, hbuf, ctr, out);
}
